// CausalSelfAttention_36481452212434
// MI455X (gfx1250) — compile-verified, matrix operand order checked by reading, not run on hardware
//
#include <hip/hip_runtime.h>
#include <hip/hip_bf16.h>

// ---------------------------------------------------------------------------
// CDNA5 (gfx1250) causal self-attention: QKV GEMM -> RoPE -> flash-attn -> O GEMM
// Matrix ops:    v_wmma_f32_16x16x32_bf16 (wave32)
// Data movement: tensor_load_to_lds (TDM, TENSORcnt) for GEMM weight tiles,
//                global_load_async_to_lds_b128 (ASYNCcnt) for activation tiles,
//                global_prefetch_b8 for attention K/V streaming.
// ---------------------------------------------------------------------------

typedef __attribute__((ext_vector_type(16))) __bf16 v16bf;
typedef __attribute__((ext_vector_type(8)))  float  v8f;
typedef __attribute__((ext_vector_type(4)))  unsigned int v4u;
typedef __attribute__((ext_vector_type(8)))  int    v8i;
typedef __attribute__((ext_vector_type(4)))  int    v4i;

#if defined(__HIP_DEVICE_COMPILE__) && __has_builtin(__builtin_amdgcn_tensor_load_to_lds)
#define HAVE_TDM 1
#else
#define HAVE_TDM 0
#endif

// ISA 7.12.2: 16-bit A operand (16x32), lane l holds row m=l&15; fragment
// element j maps to K = (j&7) + ((j>>3)<<4) + 8*(l>>4). B operand mirrors
// with lane -> column N.
__device__ __forceinline__ int kmap(int j, int g) {
  return (j & 7) + ((j >> 3) << 4) + (g << 3);
}

__device__ __forceinline__ __bf16 cvt_bf16(float x)  { return (__bf16)x; }
__device__ __forceinline__ __bf16 cvt_bf16(__bf16 x) { return x; }

__device__ __forceinline__ v8f wmma_bf16(v16bf a, v16bf b, v8f c) {
  return __builtin_amdgcn_wmma_f32_16x16x32_bf16(
      false, a, false, b, (short)0, c, false, false);
}

// Async 16B copy global -> LDS. Generic pointers to LDS carry the LDS byte
// offset in their low 32 bits.
__device__ __forceinline__ void async_copy_b128(void* lds_dst, const void* gsrc) {
  unsigned lds = (unsigned)(uintptr_t)lds_dst;
  asm volatile("global_load_async_to_lds_b128 %0, %1, off"
               :: "v"(lds), "v"(gsrc) : "memory");
}
__device__ __forceinline__ void wait_async_all() {
  asm volatile("s_wait_asynccnt 0" ::: "memory");
}

#if HAVE_TDM
// TDM 2-D tile load: rows x cols elements, row stride in elements, element
// size 4B (data_size code 2). D# per ISA 8.3/8.4; groups 2/3 zero (2-D).
__device__ __forceinline__ void tdm_load_2d_f32(unsigned lds_off, const void* gaddr,
                                                unsigned rows, unsigned cols,
                                                unsigned row_stride)
{
  const unsigned long long ga = (unsigned long long)(uintptr_t)gaddr;
  v4u g0;
  g0[0] = 1u;                                               // count=1 (valid)
  g0[1] = lds_off;                                          // lds_addr (bytes)
  g0[2] = (unsigned)(ga & 0xFFFFFFFFu);                     // global_addr lo
  g0[3] = (unsigned)((ga >> 32) & 0x1FFFFFFu) | (2u << 30); // addr hi | type=2
  v8i g1;
  g1[0] = (int)(2u << 16);                                  // data_size=4B
  g1[1] = (int)((cols & 0xFFFFu) << 16);                    // tensor_dim0 lo16
  g1[2] = (int)(((cols >> 16) & 0xFFFFu) | ((rows & 0xFFFFu) << 16)); // dim0 hi | dim1 lo
  g1[3] = (int)(((rows >> 16) & 0xFFFFu) | ((cols & 0xFFFFu) << 16)); // dim1 hi | tile_dim0
  g1[4] = (int)(rows & 0xFFFFu);                            // tile_dim1
  g1[5] = (int)row_stride;                                  // dim0_stride lo32
  g1[6] = 0;
  g1[7] = 0;
  v4i z4 = {0, 0, 0, 0};
#if __clang_major__ >= 23
  v8i z8 = {0, 0, 0, 0, 0, 0, 0, 0};
  __builtin_amdgcn_tensor_load_to_lds(g0, g1, z4, z4, z8, 0);
#else
  __builtin_amdgcn_tensor_load_to_lds(g0, g1, z4, z4, 0);
#endif
}
__device__ __forceinline__ void wait_tensor_all() {
#if __has_builtin(__builtin_amdgcn_s_wait_tensorcnt)
  __builtin_amdgcn_s_wait_tensorcnt(0);
#else
  asm volatile("s_wait_tensorcnt 0" ::: "memory");
#endif
  asm volatile("" ::: "memory");
}
#endif

// ---------------------------------------------------------------------------
// Tiled GEMM: C[M,N] f32 = A[M,K] x B[K,N]. A f32 or bf16, B f32 (weights).
// Block 256 threads (8 waves), tile 128x128, K-step 32.
// Waves in 4x2 (m x n): each wave computes 32x64 -> 8 WMMA per K-step.
// A staged via per-lane async-to-LDS; B staged via one TDM descriptor
// (wave 0) when available. bf16 conversion at fragment build.
// ---------------------------------------------------------------------------
template <typename TA>
__global__ __launch_bounds__(256)
void gemm_bf16_wmma(const TA* __restrict__ A, const float* __restrict__ B,
                    float* __restrict__ C, int M, int N, int K)
{
  __shared__ TA    sA[128 * 32];   // f32: 16 KB, bf16: 8 KB
  __shared__ float sB[32 * 128];   // 16 KB
  const int tid  = threadIdx.x;
  const int wave = tid >> 5;
  const int lane = tid & 31;
  const int g    = lane >> 4;
  const int ln   = lane & 15;
  const int m0   = blockIdx.x * 128;
  const int n0   = blockIdx.y * 128;
  const int wm   = (wave >> 1) * 32;   // wave's m strip (2 x 16-row frags)
  const int wn   = (wave & 1) * 64;    // wave's n half (4 x 16-col tiles)

  v8f acc[2][4];
  #pragma unroll
  for (int f = 0; f < 2; ++f)
    #pragma unroll
    for (int nt = 0; nt < 4; ++nt) acc[f][nt] = v8f{};

  constexpr int AROWB = 32 * (int)sizeof(TA);        // bytes per A tile row
  constexpr int ACH   = (128 * AROWB) / 16;          // 16B chunks in A tile

  for (int k0 = 0; k0 < K; k0 += 32) {
    // ---- stage A (128x32 TA) via async-to-LDS ----
    #pragma unroll
    for (int c = tid; c < ACH; c += 256) {
      const int row = (c * 16) / AROWB;
      const int bo  = (c * 16) % AROWB;
      async_copy_b128((char*)sA + c * 16,
                      (const char*)(A + (size_t)(m0 + row) * K + k0) + bo);
    }
#if HAVE_TDM
    // ---- stage B (32x128 f32) via one Tensor Data Mover descriptor ----
    if (wave == 0) {
      tdm_load_2d_f32((unsigned)(uintptr_t)&sB[0],
                      &B[(size_t)k0 * N + n0], /*rows=*/32, /*cols=*/128,
                      /*row_stride=*/(unsigned)N);
      wait_tensor_all();
    }
#else
    // ---- stage B (32x128 f32): 32 async chunks per row ----
    #pragma unroll
    for (int c = tid; c < 1024; c += 256) {
      const int row = c >> 5;
      const int co  = (c & 31) * 4;
      async_copy_b128(&sB[row * 128 + co], &B[(size_t)(k0 + row) * N + n0 + co]);
    }
#endif
    wait_async_all();
    __syncthreads();

    v16bf a0, a1;
    #pragma unroll
    for (int j = 0; j < 16; ++j) {
      const int kd = kmap(j, g);
      a0[j] = cvt_bf16(sA[(wm + ln) * 32 + kd]);
      a1[j] = cvt_bf16(sA[(wm + 16 + ln) * 32 + kd]);
    }
    #pragma unroll
    for (int nt = 0; nt < 4; ++nt) {
      v16bf bfr;
      #pragma unroll
      for (int j = 0; j < 16; ++j)
        bfr[j] = cvt_bf16(sB[kmap(j, g) * 128 + wn + nt * 16 + ln]);
      acc[0][nt] = wmma_bf16(a0, bfr, acc[0][nt]);
      acc[1][nt] = wmma_bf16(a1, bfr, acc[1][nt]);
    }
    __syncthreads();
  }

  // C/D layout: lane holds rows m = v + 8*(lane>>4), column n = lane&15.
  #pragma unroll
  for (int f = 0; f < 2; ++f)
    #pragma unroll
    for (int nt = 0; nt < 4; ++nt)
      #pragma unroll
      for (int v = 0; v < 8; ++v)
        C[(size_t)(m0 + wm + f * 16 + v + 8 * g) * N + (n0 + wn + nt * 16 + ln)] =
            acc[f][nt][v];
}

// ---------------------------------------------------------------------------
// RoPE + repack. qkv: [B*S, 3H] f32. Outputs qb,kb [B*nh,S,64] bf16 (rotated),
// vt [B*nh,64,S] bf16 (V^T). One thread per (b,h,s,pair i<32).
// ---------------------------------------------------------------------------
__global__ __launch_bounds__(256)
void rope_pack(const float* __restrict__ qkv,
               __bf16* __restrict__ qb, __bf16* __restrict__ kb,
               __bf16* __restrict__ vt)
{
  const int idx = blockIdx.x * blockDim.x + threadIdx.x;
  const int i = idx & 31;
  const int s = (idx >> 5) & 2047;
  const int h = (idx >> 16) & 15;
  const int b = idx >> 20;

  const float* row = qkv + (size_t)(b * 2048 + s) * 3072;
  const int off = h * 64 + i;

  const float inv = __powf(160000.0f, -(2.0f * (float)i) * (1.0f / 64.0f));
  const float ang = (float)s * inv;
  float sn, cs;
  __sincosf(ang, &sn, &cs);

  const float q1 = row[off],        q2 = row[off + 32];
  const float k1 = row[1024 + off], k2 = row[1024 + off + 32];
  const float v1 = row[2048 + off], v2 = row[2048 + off + 32];

  const size_t base = ((size_t)(b * 16 + h) * 2048 + s) * 64;
  qb[base + i]      = cvt_bf16(q1 * cs - q2 * sn);
  qb[base + i + 32] = cvt_bf16(q2 * cs + q1 * sn);
  kb[base + i]      = cvt_bf16(k1 * cs - k2 * sn);
  kb[base + i + 32] = cvt_bf16(k2 * cs + k1 * sn);

  const size_t vtb = (size_t)(b * 16 + h) * 64 * 2048;
  vt[vtb + (size_t)i * 2048 + s]        = cvt_bf16(v1);
  vt[vtb + (size_t)(i + 32) * 2048 + s] = cvt_bf16(v2);
}

// ---------------------------------------------------------------------------
// Flash attention (causal), bf16 WMMA, f32 online softmax in base-2 domain
// (v_exp_f32 is 2^x on AMD; 1/sqrt(64)*log2(e) folded into the score scale).
// Grid: (B*nh, S/128). Block 256 = 8 waves; wave owns 16 q-rows.
// K processed in 64-wide tiles (8 score WMMA + 8 PV WMMA per tile).
// ---------------------------------------------------------------------------
__global__ __launch_bounds__(256)
void attn_fwd(const __bf16* __restrict__ qb, const __bf16* __restrict__ kb,
              const __bf16* __restrict__ vt, __bf16* __restrict__ ob)
{
  const int bh    = blockIdx.x;
  const int wave  = threadIdx.x >> 5;
  const int lane  = threadIdx.x & 31;
  const int g     = lane >> 4;
  const int ln    = lane & 15;
  const int qrow0 = blockIdx.y * 128 + wave * 16;
  const int b     = bh >> 4;
  const int h     = bh & 15;

  const __bf16* Q  = qb + (size_t)bh * 2048 * 64;
  const __bf16* Kp = kb + (size_t)bh * 2048 * 64;
  const __bf16* Vt = vt + (size_t)bh * 64 * 2048;

  __shared__ __bf16 sP[8][16 * 64];   // per-wave P relayout staging, 16 KB
  __bf16* p = sP[wave];

  v16bf aq0, aq1;
  #pragma unroll
  for (int j = 0; j < 16; ++j) {
    const int kd = kmap(j, g);
    aq0[j] = Q[(size_t)(qrow0 + ln) * 64 + kd];
    aq1[j] = Q[(size_t)(qrow0 + ln) * 64 + kd + 32];
  }

  v8f acc[4] = {v8f{}, v8f{}, v8f{}, v8f{}};
  float mrow[8], lrow[8];
  #pragma unroll
  for (int v = 0; v < 8; ++v) { mrow[v] = -1e30f; lrow[v] = 0.0f; }

  const float qscale = 0.125f * 1.44269504f;   // 1/sqrt(64) * log2(e)
  const int kend = qrow0 + 16;                 // causal bound for this strip

  for (int kt = 0; kt < kend; kt += 64) {
    if (kt + 64 < kend) {   // prefetch next K/V tiles -> global_prefetch_b8
      __builtin_prefetch(Kp + (size_t)(kt + 64) * 64, 0, 3);
      __builtin_prefetch(Vt + (size_t)ln * 2048 + kt + 64, 0, 3);
    }

    // ---- scores: S[16q x 64k] as four 16x16 fragments ----
    v8f sfrag[4];
    #pragma unroll
    for (int ns = 0; ns < 4; ++ns) {
      v16bf bk0, bk1;
      const int kpos = kt + ns * 16 + ln;
      #pragma unroll
      for (int j = 0; j < 16; ++j) {
        const int kd = kmap(j, g);
        bk0[j] = Kp[(size_t)kpos * 64 + kd];
        bk1[j] = Kp[(size_t)kpos * 64 + kd + 32];
      }
      v8f s = v8f{};
      s = wmma_bf16(aq0, bk0, s);
      s = wmma_bf16(aq1, bk1, s);
      sfrag[ns] = s;
    }

    // ---- scale (base-2) + causal mask + tile row-max ----
    float tmax[8];
    #pragma unroll
    for (int v = 0; v < 8; ++v) tmax[v] = -1e30f;
    #pragma unroll
    for (int ns = 0; ns < 4; ++ns)
      #pragma unroll
      for (int v = 0; v < 8; ++v) {
        const int qr = qrow0 + v + 8 * g;
        float s0 = sfrag[ns][v] * qscale;
        if (kt + ns * 16 + ln > qr) s0 = -1e30f;
        sfrag[ns][v] = s0;
        tmax[v] = fmaxf(tmax[v], s0);
      }
    #pragma unroll
    for (int m = 1; m < 16; m <<= 1)
      #pragma unroll
      for (int v = 0; v < 8; ++v)
        tmax[v] = fmaxf(tmax[v], __shfl_xor(tmax[v], m, 32));

    // ---- online softmax update (2^x domain) ----
    float alpha[8], rsum[8];
    #pragma unroll
    for (int v = 0; v < 8; ++v) {
      const float mn = fmaxf(mrow[v], tmax[v]);
      alpha[v] = exp2f(mrow[v] - mn);
      mrow[v] = mn;
      float rs = 0.0f;
      #pragma unroll
      for (int ns = 0; ns < 4; ++ns) {
        const float pv = exp2f(sfrag[ns][v] - mn);
        sfrag[ns][v] = pv;
        rs += pv;
      }
      rsum[v] = rs;
    }
    #pragma unroll
    for (int m = 1; m < 16; m <<= 1)
      #pragma unroll
      for (int v = 0; v < 8; ++v)
        rsum[v] += __shfl_xor(rsum[v], m, 32);
    #pragma unroll
    for (int v = 0; v < 8; ++v) {
      lrow[v] = lrow[v] * alpha[v] + rsum[v];
      acc[0][v] *= alpha[v]; acc[1][v] *= alpha[v];
      acc[2][v] *= alpha[v]; acc[3][v] *= alpha[v];
    }

    // ---- P: C/D layout -> A-operand layout via per-wave LDS ----
    #pragma unroll
    for (int ns = 0; ns < 4; ++ns)
      #pragma unroll
      for (int v = 0; v < 8; ++v)
        p[(v + 8 * g) * 64 + ns * 16 + ln] = cvt_bf16(sfrag[ns][v]);
    asm volatile("s_wait_dscnt 0" ::: "memory");   // same-wave DS in-order
    v16bf ap0, ap1;
    #pragma unroll
    for (int j = 0; j < 16; ++j) {
      const int kd = kmap(j, g);
      ap0[j] = p[ln * 64 + kd];
      ap1[j] = p[ln * 64 + 32 + kd];
    }

    // ---- O += P x V : B-operand rows are V^T rows (contiguous in vt) ----
    #pragma unroll
    for (int nt = 0; nt < 4; ++nt) {
      v16bf bv0, bv1;
      #pragma unroll
      for (int j = 0; j < 16; ++j) {
        const int kd = kmap(j, g);
        bv0[j] = Vt[(size_t)(nt * 16 + ln) * 2048 + kt + kd];
        bv1[j] = Vt[(size_t)(nt * 16 + ln) * 2048 + kt + 32 + kd];
      }
      acc[nt] = wmma_bf16(ap0, bv0, acc[nt]);
      acc[nt] = wmma_bf16(ap1, bv1, acc[nt]);
    }
  }

  // ---- finalize: divide by row sum, write [B,S,H] bf16 ----
  float linv[8];
  #pragma unroll
  for (int v = 0; v < 8; ++v) linv[v] = 1.0f / lrow[v];
  #pragma unroll
  for (int nt = 0; nt < 4; ++nt)
    #pragma unroll
    for (int v = 0; v < 8; ++v)
      ob[(size_t)(b * 2048 + qrow0 + v + 8 * g) * 1024 + h * 64 + nt * 16 + ln] =
          cvt_bf16(acc[nt][v] * linv[v]);
}

// ---------------------------------------------------------------------------
// Launch.
// ---------------------------------------------------------------------------
extern "C" void kernel_launch(void* const* d_in, const int* in_sizes, int n_in,
                              void* d_out, int out_size, void* d_ws, size_t ws_size,
                              hipStream_t stream)
{
  (void)in_sizes; (void)n_in; (void)out_size; (void)ws_size;
  const float* hidden = (const float*)d_in[0];
  const float* Wqkv   = (const float*)d_in[2];   // d_in[1] = mask (analytic causal)
  const float* Wo     = (const float*)d_in[3];
  float* out = (float*)d_out;

  char* ws = (char*)d_ws;
  float*  qkv = (float*)ws;                                   // 4096*3072 f32
  __bf16* qb  = (__bf16*)(ws + (size_t)4096 * 3072 * 4);      // [32,2048,64]
  __bf16* kb  = qb + (size_t)32 * 2048 * 64;
  __bf16* vt  = kb + (size_t)32 * 2048 * 64;                  // [32,64,2048]
  __bf16* ao  = vt + (size_t)32 * 2048 * 64;                  // [4096,1024]

  // 1) QKV projection: [4096,1024] x [1024,3072]
  gemm_bf16_wmma<float><<<dim3(4096 / 128, 3072 / 128), 256, 0, stream>>>(
      hidden, Wqkv, qkv, 4096, 3072, 1024);

  // 2) RoPE + pack to bf16 (V transposed)
  rope_pack<<<(2u * 16 * 2048 * 32) / 256, 256, 0, stream>>>(qkv, qb, kb, vt);

  // 3) Causal flash attention
  attn_fwd<<<dim3(32, 2048 / 128), 256, 0, stream>>>(qb, kb, vt, ao);

  // 4) Output projection: [4096,1024] x [1024,1024]
  gemm_bf16_wmma<__bf16><<<dim3(4096 / 128, 1024 / 128), 256, 0, stream>>>(
      ao, Wo, out, 4096, 1024, 1024);
}
